// PopSpikeActor_11381663334672
// MI455X (gfx1250) — matrix operand order repaired and checked by reading, not compiled, not run on hardware
//
#include <hip/hip_runtime.h>
#include <hip/hip_bf16.h>
#include <stdint.h>

// ---------------------------------------------------------------------------
// PopSpikeActor forward for MI455X (gfx1250, wave32, WMMA f16->f32)
//
// B=8192, obs_dim=128, en_pop=10 (N=1280), H1=H2=256, O=80, T=5
// GEMMs run on v_wmma_f32_16x16x32_f16; LIF state update fused in epilogue.
// Per block: the shared 16-column weight tile is staged to LDS once via
// global_load_async_to_lds_b128 (ASYNCcnt) and read back with ds_load, so
// A-loads (LOADcnt), B-loads (DScnt) and WMMAs overlap on separate counters.
// Each wave register-blocks 4 M-tiles (64 batch rows x 16 neurons) giving
// 4 independent WMMA accumulator chains.
// ---------------------------------------------------------------------------

typedef __attribute__((ext_vector_type(16))) _Float16 v16h;
typedef __attribute__((ext_vector_type(8)))  _Float16 v8h;
typedef __attribute__((ext_vector_type(8)))  float    v8f;
typedef __attribute__((ext_vector_type(4)))  float    v4f;

#define BATCH   8192
#define OBSD    128
#define NPOP    1280      // obs_dim * en_pop
#define H1      256
#define H2      256
#define OOUT    80
#define TSTEPS  5
#define ENC_VTH 0.999f
#define CDECAY  0.5f
#define VDECAY  0.75f
#define VTH     0.5f

// ---- workspace layout (bytes, all offsets multiples of 256) ----------------
#define WS_SPIKES   0u                          // [B,N]  f16  20971520
#define WS_S1       20971520u                   // [B,H1] f16   4194304
#define WS_S2       25165824u                   // [B,H2] f16   4194304
#define WS_SO       29360128u                   // [B,O]  f16   1310720
#define WS_C1       30670848u                   // [B,H1] f32   8388608
#define WS_V1       39059456u
#define WS_C2       47448064u
#define WS_V2       55836672u
#define WS_CO       64225280u                   // [B,O]  f32   2621440
#define WS_VO       66846720u
#define WS_ACC      69468160u
#define WS_ZERO_BEG 20971520u
#define WS_ZERO_END 72089600u                   // zero 51118080 bytes
#define WS_W1H      72089600u                   // [H1,N] f16    655360
#define WS_W2H      72744960u                   // [H2,H1] f16   131072
#define WS_WOH      72876032u                   // [O,H2] f16     40960

union AFrag { v16h v; v8h h8[2]; };

// ---- CDNA5 async global->LDS copy (16B per lane), tracked by ASYNCcnt ------
__device__ __forceinline__ void async_copy_b128(_Float16* lds_dst,
                                                const _Float16* gsrc) {
    // VDST wants the LDS byte address: flat pointer's low 32 bits are the
    // LDS offset (ISA: LDS aperture ADDR[31:0] -> LDS offset).
    unsigned lds_addr = (unsigned)(uintptr_t)lds_dst;
    asm volatile("global_load_async_to_lds_b128 %0, %1, off"
                 :: "v"(lds_addr), "v"(gsrc)
                 : "memory");
}
__device__ __forceinline__ void async_wait0() {
    asm volatile("s_wait_asynccnt 0x0" ::: "memory");
}

// --------------------------- zero init --------------------------------------
__global__ void zero_kernel(float* __restrict__ p) {
    // exactly 51118080/16 = 3194880 float4 stores; grid sized to match
    size_t i = (size_t)blockIdx.x * blockDim.x + threadIdx.x;
    ((v4f*)p)[i] = (v4f){0.f, 0.f, 0.f, 0.f};
}

// --------------------------- weight f32 -> f16 ------------------------------
__global__ void cvt_kernel(const float* __restrict__ W1, const float* __restrict__ W2,
                           const float* __restrict__ Wo,
                           _Float16* __restrict__ o1, _Float16* __restrict__ o2,
                           _Float16* __restrict__ o3) {
    int i = blockIdx.x * blockDim.x + threadIdx.x;          // 327680 threads
    o1[i] = (_Float16)W1[i];
    if (i < H2 * H1)   o2[i] = (_Float16)W2[i];
    if (i < OOUT * H2) o3[i] = (_Float16)Wo[i];
}

// --------------------------- encoder: spikes at step t ----------------------
// Recomputes the regular-spike recurrence from scratch (t <= 4 iterations),
// so no voltage state buffer and no [T,B,N] spike train are ever stored.
__global__ void enc_step_kernel(const float* __restrict__ obs,
                                const float* __restrict__ mean,
                                const float* __restrict__ stdv,
                                _Float16* __restrict__ spikes, int t) {
    int i = blockIdx.x * blockDim.x + threadIdx.x;          // B*N threads
    int n = i % NPOP;
    int b = i / NPOP;
    int d = n / 10;                                         // en_pop = 10
    float z = (obs[b * OBSD + d] - mean[n]) / stdv[n];
    float a = __expf(-0.5f * z * z);
    float volt = 0.f, s = 0.f;
    for (int j = 0; j <= t; ++j) {
        volt += a;
        s = (volt > ENC_VTH) ? 1.f : 0.f;
        volt -= ENC_VTH * s;
    }
    spikes[i] = (_Float16)s;
}

// --------------------------- fused WMMA GEMM + LIF --------------------------
// One wave computes a 64(batch) x 16(neuron) slab of  x @ W^T  (4 M-tiles,
// 4 independent accumulators), then applies per element:
//   c = 0.5*c + gemm + bias ; v = 0.75*v*(1-s_prev) + c ; s = (v > 0.5)
// A (spike activations): row-major [B,K] f16, loaded from global.
// W: row-major [NC,K] f16; the block's 16-row tile (contiguous 16*K halves)
// is staged to LDS once with async copies, then read via ds_load.
template <int K, int NC, bool LAST>
__global__ __launch_bounds__(256)
void lif_gemm_kernel(const _Float16* __restrict__ A,
                     const _Float16* __restrict__ W,
                     const float* __restrict__ bias,
                     float* __restrict__ cst, float* __restrict__ vst,
                     _Float16* __restrict__ sst,
                     float* __restrict__ accum) {
    __shared__ __align__(64) _Float16 Bs[16 * K];   // 16 output cols x K

    const int wave = threadIdx.x >> 5;
    const int lane = threadIdx.x & 31;
    const int idx  = lane & 15;          // M (A) / N (B,C,D) within tile
    const int half = lane >> 4;          // K-half selector

    const int b0 = (blockIdx.x * 8 + wave) * 64;   // 4 M-tiles per wave
    const int h0 = blockIdx.y * 16;

    // ---- stage the block's weight tile into LDS (async, one fetch/block) ---
    {
        const _Float16* gW = W + (size_t)h0 * K;   // contiguous 16*K halves
        // 2*K chunks of 16B; 2*K is a multiple of 256 -> uniform trip count
        for (int c = threadIdx.x; c < 2 * K; c += 256)
            async_copy_b128(Bs + c * 8, gW + c * 8);
        async_wait0();
        __syncthreads();
    }

    const _Float16* __restrict__ Ar0 = A + (size_t)(b0 + idx) * K;
    const _Float16* __restrict__ Brow = Bs + (size_t)idx * K;

    v8f acc0 = {}, acc1 = {}, acc2 = {}, acc3 = {};
#pragma unroll 2
    for (int k0 = 0; k0 < K; k0 += 32) {
        // 16-bit B 32x16 layout: lane's column holds two 8-half runs:
        //   elems 0..7 -> K = k0 + 8*half + {0..7}, elems 8..15 -> K = k0 + 16 + 8*half + {0..7}   (from LDS)
        AFrag bf;
        bf.h8[0] = *(const v8h*)(Brow + k0 + 8 * half);
        bf.h8[1] = *(const v8h*)(Brow + k0 + 16 + 8 * half);
        v16h bm = bf.v;
        // 16-bit A 16x32 layout: lanes hold two contiguous 8-half runs:
        //   elems 0..7  -> K = k0 + 8*half + {0..7}
        //   elems 8..15 -> K = k0 + 16 + 8*half + {0..7}
        AFrag a0, a1, a2, a3;
        a0.h8[0] = *(const v8h*)(Ar0 + (size_t)0  * K + k0 + 8 * half);
        a0.h8[1] = *(const v8h*)(Ar0 + (size_t)0  * K + k0 + 16 + 8 * half);
        a1.h8[0] = *(const v8h*)(Ar0 + (size_t)16 * K + k0 + 8 * half);
        a1.h8[1] = *(const v8h*)(Ar0 + (size_t)16 * K + k0 + 16 + 8 * half);
        a2.h8[0] = *(const v8h*)(Ar0 + (size_t)32 * K + k0 + 8 * half);
        a2.h8[1] = *(const v8h*)(Ar0 + (size_t)32 * K + k0 + 16 + 8 * half);
        a3.h8[0] = *(const v8h*)(Ar0 + (size_t)48 * K + k0 + 8 * half);
        a3.h8[1] = *(const v8h*)(Ar0 + (size_t)48 * K + k0 + 16 + 8 * half);
        // 4 independent WMMA chains sharing one B fragment
        acc0 = __builtin_amdgcn_wmma_f32_16x16x32_f16(false, a0.v, false, bm,
                                                      (short)0, acc0, false, false);
        acc1 = __builtin_amdgcn_wmma_f32_16x16x32_f16(false, a1.v, false, bm,
                                                      (short)0, acc1, false, false);
        acc2 = __builtin_amdgcn_wmma_f32_16x16x32_f16(false, a2.v, false, bm,
                                                      (short)0, acc2, false, false);
        acc3 = __builtin_amdgcn_wmma_f32_16x16x32_f16(false, a3.v, false, bm,
                                                      (short)0, acc3, false, false);
    }

    const int   colg = h0 + idx;
    const float bb   = bias[colg];
    v8f accs[4] = {acc0, acc1, acc2, acc3};
#pragma unroll
    for (int mt = 0; mt < 4; ++mt) {
#pragma unroll
        for (int r = 0; r < 8; ++r) {
            // D layout: VGPR r holds M = r + 8*half, N = idx
            int    row = b0 + mt * 16 + r + 8 * half;
            size_t o   = (size_t)row * NC + colg;
            float  cn  = cst[o] * CDECAY + accs[mt][r] + bb;
            float  sp  = (float)sst[o];
            float  vn  = vst[o] * VDECAY * (1.0f - sp) + cn;
            float  sn  = (vn > VTH) ? 1.0f : 0.0f;
            cst[o] = cn;
            vst[o] = vn;
            sst[o] = (_Float16)sn;
            if (LAST) accum[o] += sn;
        }
    }
}

// --------------------------- decoder: grouped conv + ELU --------------------
__global__ void decode_kernel(const float* __restrict__ accO,
                              const float* __restrict__ dec_w,
                              const float* __restrict__ dec_b,
                              const float* __restrict__ log_std,
                              float* __restrict__ out) {
    int i = blockIdx.x * blockDim.x + threadIdx.x;          // B*8 threads
    int b = i >> 3, a = i & 7;
    const float* row = accO + (size_t)b * OOUT + a * 10;
    float sum = 0.f;
#pragma unroll
    for (int p = 0; p < 10; ++p) sum += row[p] * (1.0f / (float)TSTEPS) * dec_w[a * 10 + p];
    float x = sum + dec_b[a];
    out[i] = (x > 0.f) ? x : (__expf(x) - 1.f);             // ELU (alpha=1)
    if (i < 8) out[(size_t)BATCH * 8 + i] = __expf(log_std[i]);
}

// ---------------------------------------------------------------------------
extern "C" void kernel_launch(void* const* d_in, const int* in_sizes, int n_in,
                              void* d_out, int out_size, void* d_ws, size_t ws_size,
                              hipStream_t stream) {
    const float* obs      = (const float*)d_in[0];
    const float* enc_mean = (const float*)d_in[1];
    const float* enc_std  = (const float*)d_in[2];
    const float* W1       = (const float*)d_in[3];
    const float* b1       = (const float*)d_in[4];
    const float* W2       = (const float*)d_in[5];
    const float* b2       = (const float*)d_in[6];
    const float* Wo       = (const float*)d_in[7];
    const float* bo       = (const float*)d_in[8];
    const float* dec_w    = (const float*)d_in[9];
    const float* dec_b    = (const float*)d_in[10];
    const float* log_std  = (const float*)d_in[11];
    float*       out      = (float*)d_out;

    char* ws = (char*)d_ws;
    _Float16* spikes = (_Float16*)(ws + WS_SPIKES);
    _Float16* s1     = (_Float16*)(ws + WS_S1);
    _Float16* s2     = (_Float16*)(ws + WS_S2);
    _Float16* so     = (_Float16*)(ws + WS_SO);
    float*    c1     = (float*)(ws + WS_C1);
    float*    v1     = (float*)(ws + WS_V1);
    float*    c2     = (float*)(ws + WS_C2);
    float*    v2     = (float*)(ws + WS_V2);
    float*    co     = (float*)(ws + WS_CO);
    float*    vo     = (float*)(ws + WS_VO);
    float*    accO   = (float*)(ws + WS_ACC);
    _Float16* W1h    = (_Float16*)(ws + WS_W1H);
    _Float16* W2h    = (_Float16*)(ws + WS_W2H);
    _Float16* Woh    = (_Float16*)(ws + WS_WOH);

    // 1) zero all LIF state (deterministic across graph replays)
    {
        size_t nvec4 = (size_t)(WS_ZERO_END - WS_ZERO_BEG) / 16;   // 3194880
        zero_kernel<<<dim3((unsigned)(nvec4 / 256)), dim3(256), 0, stream>>>(
            (float*)(ws + WS_ZERO_BEG));
    }

    // 2) weights f32 -> f16
    cvt_kernel<<<dim3(H1 * NPOP / 256), dim3(256), 0, stream>>>(W1, W2, Wo, W1h, W2h, Woh);

    // 3) timestep loop: encoder spikes, then three fused LIF GEMMs
    //    grid.x: 8192 rows / (64 rows/wave * 8 waves/block) = 16 blocks
    for (int t = 0; t < TSTEPS; ++t) {
        enc_step_kernel<<<dim3(BATCH * NPOP / 256), dim3(256), 0, stream>>>(
            obs, enc_mean, enc_std, spikes, t);

        // layer 1: [B,1280] x [1280,256]
        lif_gemm_kernel<NPOP, H1, false>
            <<<dim3(BATCH / 64 / 8, H1 / 16), dim3(256), 0, stream>>>(
                spikes, W1h, b1, c1, v1, s1, nullptr);
        // layer 2: [B,256] x [256,256]
        lif_gemm_kernel<H1, H2, false>
            <<<dim3(BATCH / 64 / 8, H2 / 16), dim3(256), 0, stream>>>(
                s1, W2h, b2, c2, v2, s2, nullptr);
        // layer 3: [B,256] x [256,80], accumulate spikes
        lif_gemm_kernel<H2, OOUT, true>
            <<<dim3(BATCH / 64 / 8, OOUT / 16), dim3(256), 0, stream>>>(
                s2, Woh, bo, co, vo, so, accO);
    }

    // 4) decoder + std
    decode_kernel<<<dim3(BATCH * 8 / 256), dim3(256), 0, stream>>>(
        accO, dec_w, dec_b, log_std, out);
}
